// CrystalGraphConvNet_51058571215430
// MI455X (gfx1250) — matrix operand order repaired and checked by reading, not compiled, not run on hardware
//
#include <hip/hip_runtime.h>
#include <hip/hip_bf16.h>

// CGCNN forward on gfx1250 (MI455X), wave32 + WMMA bf16 16x16x32, f32 accum.
//
// Exploited identities (exact w.r.t. the reference graph):
//  * nbr_core is overwritten by nbr_filter*mask -> only first 64 output
//    channels of Wf matter (per-channel BN keeps halves independent).
//  * BN subtracts the per-channel mean -> the additive bias bf cancels.
//
// Pipeline per layer (stream-ordered, graph-capture safe, no atomics):
//   k_gemm_stats  : WMMA GEMM, per-block partial sum/sumsq of z        -> part
//   k_reduce_bn   : deterministic tree reduce -> BN1 scale/shift       -> bnp1
//   k_gemm_apply  : recompute GEMM, BN1+sigmoid^2*mask, sum over M,
//                   per-block partial BN2 stats                        -> nsum, part
//   k_reduce_bn   : BN2 scale/shift                                    -> bnp2
//   k_update      : h = softplus(h + BN2(nsum)), refresh bf16 h

typedef __attribute__((ext_vector_type(16))) __bf16       v16bf;
typedef __attribute__((ext_vector_type(8)))  float        v8f;
typedef __attribute__((ext_vector_type(4)))  unsigned int u32x4;
typedef __attribute__((ext_vector_type(2)))  unsigned int u32x2;

#define FDIM   64
#define MNBR   16
#define TWO_F  128
#define IN_DIM 192
#define ORIGF  92

union AFrag { v16bf v; u32x4 q[2]; };

// A fragment (16x32 bf16): lane = row m (lane&15), kbase=(lane>>4)*8.
// Per the ISA layout the lane holds K = [kbase..kbase+7] and [kbase+16..kbase+23].
__device__ __forceinline__ v16bf load_frag_a(const __bf16* slice, int kbase) {
  AFrag f;
  f.q[0] = *reinterpret_cast<const u32x4*>(slice + kbase);
  f.q[1] = *reinterpret_cast<const u32x4*>(slice + kbase + 16);
  return f.v;
}

// B fragment (32x16 bf16): packed per-lane contiguous (16 bf16 = 32B per lane).
__device__ __forceinline__ v16bf load_frag_b(const __bf16* tile, int lane) {
  AFrag f;
  const __bf16* p = tile + lane * 16;
  f.q[0] = *reinterpret_cast<const u32x4*>(p);
  f.q[1] = *reinterpret_cast<const u32x4*>(p + 8);
  return f.v;
}

// One wave computes z[n, 0:16, 0:64] (first half of Wf only).
// K-tiles: 0-1 self h, 2-3 gathered neighbor h, 4-5 edge features.
__device__ __forceinline__ void conv_gemm(
    const __bf16* __restrict__ hb, const __bf16* __restrict__ eb,
    const __bf16* __restrict__ bpack, const int* __restrict__ idx,
    int n, int lane, int nNodes, v8f acc[4])
{
  const int m = lane & 15;
  const int kbase = (lane >> 4) << 3;
  const int raw = idx[n * MNBR + m];
  const int j = raw < 0 ? raw + nNodes : raw;   // torch negative-index semantics
  const __bf16* hs = hb + (size_t)n * FDIM;
  const __bf16* hn = hb + (size_t)j * FDIM;
  const __bf16* er = eb + ((size_t)n * MNBR + m) * FDIM;
#pragma unroll
  for (int kt = 0; kt < 6; ++kt) {
    const __bf16* arow;
    if (kt < 2)      arow = hs + kt * 32;
    else if (kt < 4) arow = hn + (kt - 2) * 32;
    else             arow = er + (kt - 4) * 32;
    v16bf a = load_frag_a(arow, kbase);
#pragma unroll
    for (int ct = 0; ct < 4; ++ct) {
      v16bf b = load_frag_b(bpack + (kt * 4 + ct) * 512, lane);
      acc[ct] = __builtin_amdgcn_wmma_f32_16x16x32_bf16(
          false, a, false, b, (short)0, acc[ct], false, false);
    }
  }
}

// Pass 1: GEMM + per-block partial BN1 stats (sum, sumsq) over all N*M rows.
__global__ void __launch_bounds__(256)
k_gemm_stats(const __bf16* __restrict__ hb, const __bf16* __restrict__ eb,
             const __bf16* __restrict__ bpack, const int* __restrict__ idx,
             float* __restrict__ part, int nNodes)
{
  __shared__ float sm[8][128];
  const int tid = threadIdx.x, wave = tid >> 5, lane = tid & 31;
  const int n = blockIdx.x * 8 + wave;

  float sv[4] = {0.f, 0.f, 0.f, 0.f}, ssv[4] = {0.f, 0.f, 0.f, 0.f};
  if (n < nNodes) {                       // uniform per wave -> EXEC all-ones for WMMA
    v8f acc[4];
#pragma unroll
    for (int ct = 0; ct < 4; ++ct) { v8f z = {}; acc[ct] = z; }
    conv_gemm(hb, eb, bpack, idx, n, lane, nNodes, acc);
#pragma unroll
    for (int ct = 0; ct < 4; ++ct) {
      float s = 0.f, ss = 0.f;
#pragma unroll
      for (int j = 0; j < 8; ++j) { float v = acc[ct][j]; s += v; ss += v * v; }
      s  += __shfl_xor(s, 16, 32);        // combine rows held by lane and lane^16
      ss += __shfl_xor(ss, 16, 32);
      sv[ct] = s; ssv[ct] = ss;
    }
  }
#pragma unroll
  for (int ct = 0; ct < 4; ++ct) {
    if (lane < 16) {
      sm[wave][ct * 16 + lane]      = sv[ct];
      sm[wave][64 + ct * 16 + lane] = ssv[ct];
    }
  }
  __syncthreads();
  if (tid < 128) {
    float t = 0.f;
#pragma unroll
    for (int w = 0; w < 8; ++w) t += sm[w][tid];
    part[(size_t)blockIdx.x * 128 + tid] = t;
  }
}

// Pass 2: recompute GEMM, apply BN1 + sigmoid^2 * mask, sum over M neighbors,
// write nbr_sumed, emit per-block partial BN2 stats.
__global__ void __launch_bounds__(256)
k_gemm_apply(const __bf16* __restrict__ hb, const __bf16* __restrict__ eb,
             const __bf16* __restrict__ bpack, const int* __restrict__ idx,
             const float* __restrict__ bnp1, float* __restrict__ nsum,
             float* __restrict__ part, int nNodes)
{
  __shared__ float sm[8][128];
  const int tid = threadIdx.x, wave = tid >> 5, lane = tid & 31;
  const int n = blockIdx.x * 8 + wave;

  float sv[4] = {0.f, 0.f, 0.f, 0.f}, ssv[4] = {0.f, 0.f, 0.f, 0.f};
  if (n < nNodes) {
    v8f acc[4];
#pragma unroll
    for (int ct = 0; ct < 4; ++ct) { v8f z = {}; acc[ct] = z; }
    conv_gemm(hb, eb, bpack, idx, n, lane, nNodes, acc);

    // D element j in this lane is row m = j + 8*(lane>>4); build its mask bits.
    const int mhi = (lane >> 4) << 3;
    unsigned mb = 0;
#pragma unroll
    for (int j = 0; j < 8; ++j)
      if (idx[n * MNBR + mhi + j] >= 0) mb |= (1u << j);

#pragma unroll
    for (int ct = 0; ct < 4; ++ct) {
      const int c = ct * 16 + (lane & 15);
      const float sc = bnp1[c], sh = bnp1[64 + c];
      float s = 0.f;
#pragma unroll
      for (int j = 0; j < 8; ++j) {
        float z  = acc[ct][j] * sc + sh;        // BN1 (gamma/beta folded)
        float sg = 1.f / (1.f + __expf(-z));    // sigmoid
        float v  = sg * sg;                     // filter * core (core == filter)
        if (!((mb >> j) & 1u)) v = 0.f;         // invalid-neighbor mask
        s += v;
      }
      s += __shfl_xor(s, 16, 32);               // sum over all 16 neighbors
      sv[ct] = s; ssv[ct] = s * s;
      if (lane < 16) nsum[(size_t)n * FDIM + c] = s;
    }
  }
#pragma unroll
  for (int ct = 0; ct < 4; ++ct) {
    if (lane < 16) {
      sm[wave][ct * 16 + lane]      = sv[ct];
      sm[wave][64 + ct * 16 + lane] = ssv[ct];
    }
  }
  __syncthreads();
  if (tid < 128) {
    float t = 0.f;
#pragma unroll
    for (int w = 0; w < 8; ++w) t += sm[w][tid];
    part[(size_t)blockIdx.x * 128 + tid] = t;
  }
}

// Deterministic reduction of per-block partials -> BN scale/shift (64 channels).
__global__ void __launch_bounds__(1024)
k_reduce_bn(const float* __restrict__ part, int nblocks, float invCnt,
            const float* __restrict__ gamma, const float* __restrict__ beta,
            float* __restrict__ bnp)
{
  __shared__ float tot[8][128];
  const int tid = threadIdx.x;
  const int slice = tid >> 7;   // 8 slices shorten the serial add chain
  const int col = tid & 127;
  float t = 0.f;
  for (int b = slice; b < nblocks; b += 8) t += part[(size_t)b * 128 + col];
  tot[slice][col] = t;
  __syncthreads();
  if (tid < 128) {
    float u = 0.f;
#pragma unroll
    for (int s = 0; s < 8; ++s) u += tot[s][tid];
    tot[0][tid] = u;
  }
  __syncthreads();
  if (tid < 64) {
    float mean = tot[0][tid] * invCnt;
    float var  = fmaxf(tot[0][64 + tid] * invCnt - mean * mean, 0.f);
    float r    = rsqrtf(var + 1e-5f);
    float sc   = gamma[tid] * r;
    bnp[tid]      = sc;
    bnp[64 + tid] = beta[tid] - mean * sc;
  }
}

// h = softplus(h + BN2(nbr_sumed)); refresh bf16 mirror; optionally emit output.
__global__ void __launch_bounds__(256)
k_update(float* __restrict__ h, const float* __restrict__ nsum,
         const float* __restrict__ bnp2, __bf16* __restrict__ hb,
         float* __restrict__ out, int total)
{
  int i = blockIdx.x * 256 + threadIdx.x;
  if (i >= total) return;
  int c = i & (FDIM - 1);
  float x = h[i] + nsum[i] * bnp2[c] + bnp2[64 + c];
  float y = (x > 30.f) ? x : log1pf(__expf(x));
  h[i]  = y;
  hb[i] = (__bf16)y;
  if (out) out[i] = y;
}

// Embedding: h = node_fea @ W_emb + b_emb (0.6 GFLOP, negligible; plain VALU).
__global__ void __launch_bounds__(64)
k_embed(const float* __restrict__ nf, const float* __restrict__ W,
        const float* __restrict__ b, float* __restrict__ h,
        __bf16* __restrict__ hb, int nNodes)
{
  __shared__ float row[ORIGF];
  const int n = blockIdx.x, c = threadIdx.x;
  if (n >= nNodes) return;
  for (int k = c; k < ORIGF; k += 64) row[k] = nf[(size_t)n * ORIGF + k];
  __syncthreads();
  float acc = b[c];
#pragma unroll 4
  for (int k = 0; k < ORIGF; ++k) acc = fmaf(row[k], W[k * FDIM + c], acc);
  h[(size_t)n * FDIM + c]  = acc;
  hb[(size_t)n * FDIM + c] = (__bf16)acc;
}

// One-time: edge features f32 -> bf16 (halves the dominant HBM stream).
__global__ void __launch_bounds__(256)
k_cvt_edge(const float* __restrict__ e, __bf16* __restrict__ eb, long n4)
{
  long i = (long)blockIdx.x * 256 + threadIdx.x;
  if (i >= n4) return;
  const float4 f = reinterpret_cast<const float4*>(e)[i];
  union { __bf16 h[4]; u32x2 u; } o;
  o.h[0] = (__bf16)f.x; o.h[1] = (__bf16)f.y;
  o.h[2] = (__bf16)f.z; o.h[3] = (__bf16)f.w;
  *reinterpret_cast<u32x2*>(eb + i * 4) = o.u;
}

// One-time: pack Wf[:, :64] (bf16) into per-lane WMMA B fragments.
// Layout: [(kt*4+ct)*512 + lane*16 + t] = Wf[kt*32 + (lane>>4)*8 + (t&7) + (t>>3)*16][ct*16 + (lane&15)]
__global__ void __launch_bounds__(256)
k_pack_wf(const float* __restrict__ Wf, __bf16* __restrict__ bp)
{
  int t = blockIdx.x * 256 + threadIdx.x;
  if (t >= 3 * 12288) return;
  int i = t / 12288, r = t % 12288;
  int tile = r >> 9, within = r & 511;
  int lane = within >> 4, tt = within & 15;
  int kt = tile >> 2, ct = tile & 3;
  int col = ct * 16 + (lane & 15);
  int k   = kt * 32 + ((lane >> 4) << 3) + (tt & 7) + ((tt >> 3) << 4);
  bp[t] = (__bf16)Wf[((size_t)i * IN_DIM + k) * TWO_F + col];
}

extern "C" void kernel_launch(void* const* d_in, const int* in_sizes, int n_in,
                              void* d_out, int out_size, void* d_ws, size_t ws_size,
                              hipStream_t stream) {
  (void)n_in; (void)out_size; (void)ws_size;
  const float* node_fea = (const float*)d_in[0];
  const float* edge_fea = (const float*)d_in[1];
  const int*   eidx     = (const int*)d_in[2];
  const float* W_emb    = (const float*)d_in[3];
  const float* b_emb    = (const float*)d_in[4];
  const float* Wf       = (const float*)d_in[5];
  // d_in[6] (bf) cancels under BN's mean subtraction.
  const float* g1  = (const float*)d_in[7];
  const float* be1 = (const float*)d_in[8];
  const float* g2  = (const float*)d_in[9];
  const float* be2 = (const float*)d_in[10];

  const int nNodes  = in_sizes[2] / MNBR;
  const int nblocks = (nNodes + 7) / 8;

  char* p = (char*)d_ws;
  auto carve = [&](size_t bytes) {
    char* q = p; p += (bytes + 255) & ~(size_t)255; return q;
  };
  float*  h     = (float*)carve((size_t)nNodes * FDIM * 4);
  float*  nsum  = (float*)carve((size_t)nNodes * FDIM * 4);
  float*  part  = (float*)carve((size_t)nblocks * 128 * 4);
  float*  bnp1  = (float*)carve(128 * 4);
  float*  bnp2  = (float*)carve(128 * 4);
  __bf16* hb    = (__bf16*)carve((size_t)nNodes * FDIM * 2);
  __bf16* eb    = (__bf16*)carve((size_t)nNodes * MNBR * FDIM * 2);
  __bf16* bpack = (__bf16*)carve((size_t)3 * 12288 * 2);

  k_embed<<<nNodes, 64, 0, stream>>>(node_fea, W_emb, b_emb, h, hb, nNodes);
  const long n4 = (long)nNodes * MNBR * FDIM / 4;
  k_cvt_edge<<<(int)((n4 + 255) / 256), 256, 0, stream>>>(edge_fea, eb, n4);
  k_pack_wf<<<(3 * 12288 + 255) / 256, 256, 0, stream>>>(Wf, bpack);

  const float invE = 1.f / (float)((size_t)nNodes * MNBR);
  const float invN = 1.f / (float)nNodes;
  for (int i = 0; i < 3; ++i) {
    const __bf16* bp = bpack + (size_t)i * 12288;
    k_gemm_stats<<<nblocks, 256, 0, stream>>>(hb, eb, bp, eidx, part, nNodes);
    k_reduce_bn<<<1, 1024, 0, stream>>>(part, nblocks, invE,
                                        g1 + i * TWO_F, be1 + i * TWO_F, bnp1);
    k_gemm_apply<<<nblocks, 256, 0, stream>>>(hb, eb, bp, eidx, bnp1, nsum, part, nNodes);
    k_reduce_bn<<<1, 1024, 0, stream>>>(part, nblocks, invN,
                                        g2 + i * FDIM, be2 + i * FDIM, bnp2);
    k_update<<<(nNodes * FDIM + 255) / 256, 256, 0, stream>>>(
        h, nsum, bnp2, hb, (i == 2) ? (float*)d_out : nullptr, nNodes * FDIM);
  }
}
